// SelfAttention2d_37374805409948
// MI455X (gfx1250) — compile-verified, matrix operand order checked by reading, not run on hardware
//
#include <hip/hip_runtime.h>
#include <hip/hip_bf16.h>

// ---------------------------------------------------------------------------
// SelfAttention2d on gfx1250 (CDNA5): GroupNorm -> QKV WMMA GEMM ->
// flash-attention (WMMA QK^T + PV, online softmax) -> proj WMMA GEMM + residual
// b=4, C=512, H=W=48 -> s=2304, heads=8, hd=64, groups=32
// GEMMs and attention are software-pipelined (register / LDS double buffering)
// so fragment loads overlap WMMA issue instead of full s_wait_loadcnt drains.
// ---------------------------------------------------------------------------

typedef __attribute__((ext_vector_type(16))) _Float16 v16h;
typedef __attribute__((ext_vector_type(8)))  float    v8f;

#define BATCH 4
#define CCH   512
#define SEQ   2304
#define HEADS 8
#define HD    64
#define NQKV  1536
#define GROUPS 32
#define GSZ   (16 * SEQ)    /* elems per (b,group) = 36864 */
#define MT_PER_B (SEQ / 16) /* 144 M-tiles per batch image */

union FragU {
    v16h  v;
    uint4 u[2];
};

// Load one 16x32 f16 WMMA fragment (A layout; B uses same layout with N as the
// "row" dim when stored N-major). Per CDNA5 ISA 7.12.2 (16-bit A 16x32):
//   lane<16 : row=lane,    K in {0..7, 16..23}
//   lane>=16: row=lane-16, K in {8..15, 24..31}
// i.e. two 16-byte chunks at (row*ld + hi*8) and (+16).
static __device__ __forceinline__ v16h load_frag(const _Float16* base, int ld, int lane) {
    const int r  = lane & 15;
    const int hi = (lane >> 4) & 1;
    const _Float16* p = base + r * ld + hi * 8;
    FragU f;
    f.u[0] = *(const uint4*)(p);
    f.u[1] = *(const uint4*)(p + 16);
    return f.v;
}

static __device__ __forceinline__ v8f wmma16(v16h a, v16h b, v8f c) {
    return __builtin_amdgcn_wmma_f32_16x16x32_f16(
        /*neg_a=*/false, a, /*neg_b=*/false, b,
        /*c_mod=*/(short)0, c, /*reuse_a=*/false, /*reuse_b=*/false);
}

static __device__ __forceinline__ float hred_max(float x) {
    #pragma unroll
    for (int m = 8; m > 0; m >>= 1) x = fmaxf(x, __shfl_xor(x, m, 16));
    return x;
}
static __device__ __forceinline__ float hred_sum(float x) {
    #pragma unroll
    for (int m = 8; m > 0; m >>= 1) x += __shfl_xor(x, m, 16);
    return x;
}

// ---------------------------------------------------------------------------
// GroupNorm pass 1: per (b,group) mean & rstd.  grid=128 blocks of 256.
// ---------------------------------------------------------------------------
__global__ void __launch_bounds__(256) gn_stats(const float* __restrict__ x,
                                                float* __restrict__ stats) {
    const int bg  = blockIdx.x;                 // b*32 + g
    const int tid = threadIdx.x;
    const float* p = x + (size_t)bg * GSZ;
    float s = 0.f, ss = 0.f;
    for (int i = tid; i < GSZ; i += 256) {
        float v = p[i];
        s += v; ss += v * v;
    }
    __shared__ float sh[512];
    sh[tid] = s; sh[256 + tid] = ss;
    __syncthreads();
    for (int off = 128; off > 0; off >>= 1) {
        if (tid < off) {
            sh[tid]       += sh[tid + off];
            sh[256 + tid] += sh[256 + tid + off];
        }
        __syncthreads();
    }
    if (tid == 0) {
        float mean = sh[0] / (float)GSZ;
        float var  = sh[256] / (float)GSZ - mean * mean;
        stats[bg * 2 + 0] = mean;
        stats[bg * 2 + 1] = rsqrtf(var + 1e-5f);
    }
}

// ---------------------------------------------------------------------------
// GroupNorm pass 2: normalize + affine, write xn as (b, s, c) f16 (GEMM-A).
// ---------------------------------------------------------------------------
__global__ void __launch_bounds__(256) gn_apply(const float* __restrict__ x,
                                                const float* __restrict__ gamma,
                                                const float* __restrict__ beta,
                                                const float* __restrict__ stats,
                                                _Float16* __restrict__ xn) {
    int idx = blockIdx.x * 256 + threadIdx.x;
    const int total = BATCH * CCH * SEQ;
    if (idx >= total) return;
    int s = idx % SEQ;
    int c = (idx / SEQ) % CCH;
    int b = idx / (SEQ * CCH);
    int g = c >> 4;                               // 16 channels per group
    float mean = stats[(b * GROUPS + g) * 2 + 0];
    float rstd = stats[(b * GROUPS + g) * 2 + 1];
    float v = (x[idx] - mean) * rstd * gamma[c] + beta[c];
    xn[((size_t)b * SEQ + s) * CCH + c] = (_Float16)v;
}

// ---------------------------------------------------------------------------
// Weight convert+transpose: w (K,N) f32 row-major -> wT (N,K) f16.
// ---------------------------------------------------------------------------
__global__ void __launch_bounds__(256) w_transpose(const float* __restrict__ w,
                                                   _Float16* __restrict__ wT,
                                                   int K, int N) {
    int idx = blockIdx.x * 256 + threadIdx.x;
    if (idx >= K * N) return;
    int n = idx % N;
    int k = idx / N;
    wT[(size_t)n * K + k] = (_Float16)w[idx];
}

// ---------------------------------------------------------------------------
// Pipelined 16Mx64N GEMM core: A (M-major, ld=CCH), B0..B3 (N-major, ld=CCH).
// Next iteration's fragments are fetched before the current WMMAs consume the
// previous set, so loads overlap matrix issue (partial loadcnt waits).
// ---------------------------------------------------------------------------
static __device__ __forceinline__ void gemm_16x64(const _Float16* A,
                                                  const _Float16* B0,
                                                  const _Float16* B1,
                                                  const _Float16* B2,
                                                  const _Float16* B3,
                                                  int lane,
                                                  v8f& acc0, v8f& acc1,
                                                  v8f& acc2, v8f& acc3) {
    v16h af  = load_frag(A,  CCH, lane);
    v16h bf0 = load_frag(B0, CCH, lane);
    v16h bf1 = load_frag(B1, CCH, lane);
    v16h bf2 = load_frag(B2, CCH, lane);
    v16h bf3 = load_frag(B3, CCH, lane);
    for (int kk = 32; kk < CCH; kk += 32) {
        v16h an  = load_frag(A  + kk, CCH, lane);
        v16h bn0 = load_frag(B0 + kk, CCH, lane);
        v16h bn1 = load_frag(B1 + kk, CCH, lane);
        v16h bn2 = load_frag(B2 + kk, CCH, lane);
        v16h bn3 = load_frag(B3 + kk, CCH, lane);
        acc0 = wmma16(af, bf0, acc0);
        acc1 = wmma16(af, bf1, acc1);
        acc2 = wmma16(af, bf2, acc2);
        acc3 = wmma16(af, bf3, acc3);
        af = an; bf0 = bn0; bf1 = bn1; bf2 = bn2; bf3 = bn3;
    }
    acc0 = wmma16(af, bf0, acc0);
    acc1 = wmma16(af, bf1, acc1);
    acc2 = wmma16(af, bf2, acc2);
    acc3 = wmma16(af, bf3, acc3);
}

// ---------------------------------------------------------------------------
// QKV GEMM: (9216 x 512) @ (512 x 1536) + bias, outputs q/k/v as
// (b, head, s, hd) f16.  grid=(576, 6), block=128 (4 waves); one 16Mx64N
// strip per wave.
// ---------------------------------------------------------------------------
__global__ void __launch_bounds__(128) qkv_gemm(const _Float16* __restrict__ xn,
                                                const _Float16* __restrict__ wT,
                                                const float* __restrict__ bias,
                                                _Float16* __restrict__ q,
                                                _Float16* __restrict__ k,
                                                _Float16* __restrict__ v) {
    const int lane = threadIdx.x & 31;
    const int wave = threadIdx.x >> 5;
    const int mt   = blockIdx.x;                  // 0..575
    const int n0   = (blockIdx.y * 4 + wave) * 64;

    const _Float16* A = xn + (size_t)mt * 16 * CCH;
    const _Float16* B = wT + (size_t)n0 * CCH;

    v8f acc0 = {}, acc1 = {}, acc2 = {}, acc3 = {};
    gemm_16x64(A, B, B + 16 * CCH, B + 32 * CCH, B + 48 * CCH, lane,
               acc0, acc1, acc2, acc3);

    // 16-row M-tile lies entirely inside one batch (2304 % 16 == 0)
    const int b     = mt / MT_PER_B;
    const int sbase = (mt % MT_PER_B) * 16;
    const int hi    = (lane >> 4) & 1;
    const int col   = lane & 15;

    const v8f accs[4] = {acc0, acc1, acc2, acc3};
    #pragma unroll
    for (int t = 0; t < 4; ++t) {
        const int n     = n0 + t * 16 + col;
        const int which = n >> 9;                 // 0=q 1=k 2=v (tile never straddles)
        const int cl    = n & 511;
        const int head  = cl >> 6;
        const int d     = cl & 63;
        _Float16* dst = ((which == 0) ? q : (which == 1) ? k : v)
                      + ((((size_t)b * HEADS + head) * SEQ) + sbase) * HD + d;
        const float bv = bias[n];
        #pragma unroll
        for (int i = 0; i < 8; ++i) {
            dst[(size_t)(i + hi * 8) * HD] = (_Float16)(accs[t][i] + bv);
        }
    }
}

// ---------------------------------------------------------------------------
// Flash attention: grid=(36, 8, 4), block=128 (4 waves).  Each wave owns 16
// query rows; the block double-buffers 32-key K / V^T LDS tiles so global
// staging overlaps WMMA + softmax, with a single barrier per iteration.
// ---------------------------------------------------------------------------
__global__ void __launch_bounds__(128) attn(const _Float16* __restrict__ q,
                                            const _Float16* __restrict__ k,
                                            const _Float16* __restrict__ v,
                                            _Float16* __restrict__ ao) {
    __shared__ _Float16 sK[2][32][64];            // [buf][key][d]
    __shared__ _Float16 sVt[2][64][32];           // [buf][d][key] (transposed)
    __shared__ _Float16 sP[4][16][32];            // per-wave P restage

    const int lane  = threadIdx.x & 31;
    const int wave  = threadIdx.x >> 5;
    const int head  = blockIdx.y;
    const int batch = blockIdx.z;
    const int qrow0 = blockIdx.x * 64 + wave * 16;

    const size_t bh = (size_t)batch * HEADS + head;
    const _Float16* Q  = q + bh * SEQ * HD;
    const _Float16* Kp = k + bh * SEQ * HD;
    const _Float16* Vp = v + bh * SEQ * HD;

    const v16h qf0 = load_frag(Q + (size_t)qrow0 * HD + 0,  HD, lane);
    const v16h qf1 = load_frag(Q + (size_t)qrow0 * HD + 32, HD, lane);

    v8f o0 = {}, o1 = {}, o2 = {}, o3 = {};
    float mrow[8], lrow[8];
    #pragma unroll
    for (int i = 0; i < 8; ++i) { mrow[i] = -1e30f; lrow[i] = 0.f; }

    const int ldr  = threadIdx.x >> 2;            // 0..31 (key row)
    const int ldc  = (threadIdx.x & 3) * 16;      // 0/16/32/48
    const int hi   = (lane >> 4) & 1;
    const int col  = lane & 15;
    const float scale = 0.125f;                   // 64^-0.5

    uint4 kr0, kr1, vr0, vr1;
    // preload tile 0 into registers, stage to buffer 0
    {
        const _Float16* kp = Kp + (size_t)ldr * HD + ldc;
        kr0 = *(const uint4*)(kp + 0);
        kr1 = *(const uint4*)(kp + 8);
        const _Float16* vp = Vp + (size_t)ldr * HD + ldc;
        vr0 = *(const uint4*)(vp + 0);
        vr1 = *(const uint4*)(vp + 8);
        *(uint4*)(&sK[0][ldr][ldc + 0]) = kr0;
        *(uint4*)(&sK[0][ldr][ldc + 8]) = kr1;
        FragU vf; vf.u[0] = vr0; vf.u[1] = vr1;
        const _Float16* vh = (const _Float16*)&vf;
        #pragma unroll
        for (int j = 0; j < 16; ++j) sVt[0][ldc + j][ldr] = vh[j];
    }
    __syncthreads();

    int cur = 0;
    for (int kt = 0; kt < SEQ; kt += 32) {
        const bool has_next = (kt + 32) < SEQ;
        // ---- prefetch next tile into registers (latency hidden by compute) ----
        if (has_next) {
            const _Float16* kp = Kp + (size_t)(kt + 32 + ldr) * HD + ldc;
            kr0 = *(const uint4*)(kp + 0);
            kr1 = *(const uint4*)(kp + 8);
            const _Float16* vp = Vp + (size_t)(kt + 32 + ldr) * HD + ldc;
            vr0 = *(const uint4*)(vp + 0);
            vr1 = *(const uint4*)(vp + 8);
        }

        // ---- S = Q K^T (two 16-key tiles, K-dim 64 = 2 wmma each) ----
        v8f s0 = {}, s1 = {};
        s0 = wmma16(qf0, load_frag(&sK[cur][0][0],   64, lane), s0);
        s0 = wmma16(qf1, load_frag(&sK[cur][0][32],  64, lane), s0);
        s1 = wmma16(qf0, load_frag(&sK[cur][16][0],  64, lane), s1);
        s1 = wmma16(qf1, load_frag(&sK[cur][16][32], 64, lane), s1);

        // ---- online softmax (rows i / i+8 live in lane halves) ----
        #pragma unroll
        for (int i = 0; i < 8; ++i) {
            float a0 = (float)s0[i] * scale;
            float a1 = (float)s1[i] * scale;
            float mx = hred_max(fmaxf(a0, a1));
            float mnew = fmaxf(mrow[i], mx);
            float alpha = __expf(mrow[i] - mnew);
            float p0 = __expf(a0 - mnew);
            float p1 = __expf(a1 - mnew);
            lrow[i] = lrow[i] * alpha + hred_sum(p0 + p1);
            mrow[i] = mnew;
            o0[i] *= alpha; o1[i] *= alpha; o2[i] *= alpha; o3[i] *= alpha;
            int r = i + hi * 8;
            sP[wave][r][col]      = (_Float16)p0;
            sP[wave][r][16 + col] = (_Float16)p1;
        }
        asm volatile("s_wait_dscnt 0" ::: "memory");  // order P stores vs reads (same wave)

        // ---- O += P V  (K-dim = 32 keys, N = d in 4 tiles of 16) ----
        {
            v16h pf = load_frag(&sP[wave][0][0], 32, lane);
            o0 = wmma16(pf, load_frag(&sVt[cur][0][0],  32, lane), o0);
            o1 = wmma16(pf, load_frag(&sVt[cur][16][0], 32, lane), o1);
            o2 = wmma16(pf, load_frag(&sVt[cur][32][0], 32, lane), o2);
            o3 = wmma16(pf, load_frag(&sVt[cur][48][0], 32, lane), o3);
        }

        // ---- stage next tile to the alternate buffer; safe because it was
        //      last read in iteration i-1 (barrier there covers all waves) ----
        if (has_next) {
            const int nxt = cur ^ 1;
            *(uint4*)(&sK[nxt][ldr][ldc + 0]) = kr0;
            *(uint4*)(&sK[nxt][ldr][ldc + 8]) = kr1;
            FragU vf; vf.u[0] = vr0; vf.u[1] = vr1;
            const _Float16* vh = (const _Float16*)&vf;
            #pragma unroll
            for (int j = 0; j < 16; ++j) sVt[nxt][ldc + j][ldr] = vh[j];
        }
        __syncthreads();
        cur ^= 1;
    }

    // ---- normalize + store attn_out (b, s, c) f16 ----
    #pragma unroll
    for (int i = 0; i < 8; ++i) {
        float inv = 1.0f / lrow[i];
        int row = qrow0 + i + hi * 8;
        _Float16* dst = ao + ((size_t)batch * SEQ + row) * CCH + head * HD;
        dst[0 * 16 + col] = (_Float16)(o0[i] * inv);
        dst[1 * 16 + col] = (_Float16)(o1[i] * inv);
        dst[2 * 16 + col] = (_Float16)(o2[i] * inv);
        dst[3 * 16 + col] = (_Float16)(o3[i] * inv);
    }
}

// ---------------------------------------------------------------------------
// Output projection + residual: out(b,c,h,w) = x + attn_out @ w_proj + b_proj
// grid=(576, 2), block=128; one 16Mx64N strip per wave.
// ---------------------------------------------------------------------------
__global__ void __launch_bounds__(128) proj_gemm(const _Float16* __restrict__ ao,
                                                 const _Float16* __restrict__ wT,
                                                 const float* __restrict__ bias,
                                                 const float* __restrict__ x,
                                                 float* __restrict__ out) {
    const int lane = threadIdx.x & 31;
    const int wave = threadIdx.x >> 5;
    const int mt   = blockIdx.x;
    const int n0   = (blockIdx.y * 4 + wave) * 64;

    const _Float16* A = ao + (size_t)mt * 16 * CCH;
    const _Float16* B = wT + (size_t)n0 * CCH;

    v8f acc0 = {}, acc1 = {}, acc2 = {}, acc3 = {};
    gemm_16x64(A, B, B + 16 * CCH, B + 32 * CCH, B + 48 * CCH, lane,
               acc0, acc1, acc2, acc3);

    const int b     = mt / MT_PER_B;
    const int sbase = (mt % MT_PER_B) * 16;
    const int hi    = (lane >> 4) & 1;
    const int col   = lane & 15;

    const v8f accs[4] = {acc0, acc1, acc2, acc3};
    #pragma unroll
    for (int t = 0; t < 4; ++t) {
        const int n  = n0 + t * 16 + col;
        const float bv = bias[n];
        const float* xs = x   + ((size_t)b * CCH + n) * SEQ + sbase;
        float*       os = out + ((size_t)b * CCH + n) * SEQ + sbase;
        #pragma unroll
        for (int i = 0; i < 8; ++i) {
            int s = i + hi * 8;
            os[s] = xs[s] + accs[t][i] + bv;
        }
    }
}

// ---------------------------------------------------------------------------
extern "C" void kernel_launch(void* const* d_in, const int* in_sizes, int n_in,
                              void* d_out, int out_size, void* d_ws, size_t ws_size,
                              hipStream_t stream) {
    const float* x      = (const float*)d_in[0];   // (4,512,48,48)
    const float* gamma  = (const float*)d_in[1];   // (512,)
    const float* beta   = (const float*)d_in[2];   // (512,)
    const float* w_qkv  = (const float*)d_in[3];   // (512,1536)
    const float* b_qkv  = (const float*)d_in[4];   // (1536,)
    const float* w_proj = (const float*)d_in[5];   // (512,512)
    const float* b_proj = (const float*)d_in[6];   // (512,)
    float* out = (float*)d_out;                    // (4,512,48,48)

    char* ws = (char*)d_ws;
    size_t off = 0;
    auto carve = [&](size_t bytes) -> char* {
        char* p = ws + off;
        off = (off + bytes + 255) & ~(size_t)255;
        return p;
    };
    float*    stats  = (float*)   carve(BATCH * GROUPS * 2 * sizeof(float));
    _Float16* xn     = (_Float16*)carve((size_t)BATCH * SEQ * CCH * 2);
    _Float16* wqkvT  = (_Float16*)carve((size_t)NQKV * CCH * 2);
    _Float16* wprojT = (_Float16*)carve((size_t)CCH * CCH * 2);
    _Float16* qb     = (_Float16*)carve((size_t)BATCH * HEADS * SEQ * HD * 2);
    _Float16* kb     = (_Float16*)carve((size_t)BATCH * HEADS * SEQ * HD * 2);
    _Float16* vb     = (_Float16*)carve((size_t)BATCH * HEADS * SEQ * HD * 2);
    _Float16* aob    = (_Float16*)carve((size_t)BATCH * SEQ * CCH * 2);
    (void)ws_size; (void)n_in; (void)in_sizes; (void)out_size;

    // 1. GroupNorm
    gn_stats<<<BATCH * GROUPS, 256, 0, stream>>>(x, stats);
    {
        int total = BATCH * CCH * SEQ;
        gn_apply<<<(total + 255) / 256, 256, 0, stream>>>(x, gamma, beta, stats, xn);
    }
    // 2. Weight transposes (f32 -> f16, N-major)
    w_transpose<<<(CCH * NQKV + 255) / 256, 256, 0, stream>>>(w_qkv, wqkvT, CCH, NQKV);
    w_transpose<<<(CCH * CCH + 255) / 256, 256, 0, stream>>>(w_proj, wprojT, CCH, CCH);
    // 3. QKV GEMM (pipelined 16x64 strip per wave)
    qkv_gemm<<<dim3(BATCH * SEQ / 16, NQKV / 64 / 4), 128, 0, stream>>>(
        xn, wqkvT, b_qkv, qb, kb, vb);
    // 4. Flash attention (double-buffered LDS tiles)
    attn<<<dim3(SEQ / 64, HEADS, BATCH), 128, 0, stream>>>(qb, kb, vb, aob);
    // 5. Projection + residual (pipelined 16x64 strip per wave)
    proj_gemm<<<dim3(BATCH * SEQ / 16, CCH / 64 / 4), 128, 0, stream>>>(
        aob, wprojT, b_proj, x, out);
}
